// AICL_5291399708975
// MI455X (gfx1250) — matrix operand order repaired and checked by reading, not compiled, not run on hardware
//
#include <hip/hip_runtime.h>
#include <hip/hip_bf16.h>
#include <math.h>

// ---------------------------------------------------------------------------
// AICL forward for MI455X (gfx1250, wave32, WMMA + TDM).
// Core: single fused bf16 GEMM (M=B*T=32768, N=2560, K=3*2048) implementing
// all five K=3 conv1ds via folded weights. v_wmma_f32_16x16x32_bf16, LDS
// double buffering; B tiles staged by the Tensor Data Mover (with LDS pad to
// keep the bank-conflict-free 96B row stride), A tiles converted f32->bf16 in
// VGPRs and stored as b128. Compute-bound (~1 TFLOP vs ~0.6GB @ 23.3 TB/s).
// ---------------------------------------------------------------------------

typedef __attribute__((ext_vector_type(16))) __bf16 v16bf;
typedef __attribute__((ext_vector_type(8)))  float  v8f;
typedef unsigned int u32x4_ __attribute__((ext_vector_type(4)));
typedef int          i32x4_ __attribute__((ext_vector_type(4)));
typedef int          i32x8_ __attribute__((ext_vector_type(8)));

#define B_    16
#define T_    2048
#define CIN_  2048
#define NOUT_ 2560
#define NCLS  20
#define KC_   102     // T//20
#define KI_   68      // T//30
#define KC_TOP 81     // int(102*0.8)
#define KC_RND 21
#define KI_TOP 54     // int(68*0.8)
#define KI_RND 14

// output offsets (in floats), return-tuple order of the reference
#define OFF_CAS    0
#define OFF_AFLOW  655360
#define OFF_ARGB   688128
#define OFF_MIN    720896
#define SRC_STRIDE 2785280
#define CAT1_OFF   835584
#define CAT2_OFF   1671168
#define CAT3_OFF   2228224
#define OFF_AN1    9076736
#define OFF_AN2    9109504
#define OFF_BIN1   9142272
#define OFF_BIN2   9175040
#define OFF_GATE   9207808

// ------------------------------ helpers ------------------------------------

__device__ __forceinline__ float sigmoidf_(float x) { return 1.f / (1.f + __expf(-x)); }

__device__ __forceinline__ float rnd01_(unsigned a, unsigned b, unsigned c, unsigned d) {
  unsigned h = a * 0x9E3779B1u ^ b * 0x85EBCA77u ^ c * 0xC2B2AE3Du ^ d * 0x27D4EB2Fu;
  h ^= h >> 16; h *= 0x7FEB352Du; h ^= h >> 15; h *= 0x846CA68Bu; h ^= h >> 16;
  return (float)(h >> 8) * (1.0f / 16777216.0f);
}

// pack two f32 into packed bf16 (round-to-nearest-even)
__device__ __forceinline__ unsigned pkbf_(float a, float b) {
  unsigned ua = __float_as_uint(a);
  unsigned ub = __float_as_uint(b);
  ua = (ua + 0x7FFFu + ((ua >> 16) & 1u)) >> 16;
  ub = (ub + 0x7FFFu + ((ub >> 16) & 1u)) & 0xFFFF0000u;
  return ua | ub;
}

__device__ __forceinline__ float blockReduceSum(float v, float* red) {
  const int tid = threadIdx.x;
  red[tid] = v;
  __syncthreads();
  for (int s = 128; s > 0; s >>= 1) {
    if (tid < s) red[tid] += red[tid + s];
    __syncthreads();
  }
  float r = red[0];
  __syncthreads();
  return r;
}

__device__ void bitonicDesc_(float* sv, int* si) {
  for (int k = 2; k <= T_; k <<= 1) {
    for (int j = k >> 1; j > 0; j >>= 1) {
      for (int i = threadIdx.x; i < T_; i += 256) {
        int ixj = i ^ j;
        if (ixj > i) {
          bool desc = ((i & k) == 0);
          float a = sv[i], c = sv[ixj];
          bool sw = desc ? (a < c) : (a > c);
          if (sw) {
            sv[i] = c; sv[ixj] = a;
            int t = si[i]; si[i] = si[ixj]; si[ixj] = t;
          }
        }
      }
      __syncthreads();
    }
  }
}

// --------------------------- weight packing --------------------------------
// WBt[tap][kTile(64)][n(2560)][ki(32)]  bf16 : each (tap,kTile) B-panel is a
// contiguous [2560][32] block; a 128-row N slice is one contiguous 8KB tile.
__global__ void aicl_pack_w(const float* __restrict__ w_rgb, const float* __restrict__ w_flow,
                            const float* __restrict__ w_m1,  const float* __restrict__ w_m2,
                            const float* __restrict__ w_g1,  __bf16* __restrict__ WBt) {
  const int total = 3 * CIN_ * NOUT_;
  for (int idx = blockIdx.x * blockDim.x + threadIdx.x; idx < total;
       idx += gridDim.x * blockDim.x) {
    int ki = idx & 31;
    int n  = (idx >> 5) % NOUT_;
    int ct = (idx / (32 * NOUT_)) % 64;
    int k  = idx / (32 * NOUT_ * 64);
    int ci = ct * 32 + ki;
    float v = 0.f;
    if (n < 512) {
      if (ci < 1024) v = w_rgb[(n * 1024 + ci) * 3 + k];
    } else if (n < 1024) {
      if (ci >= 1024) v = w_flow[((n - 512) * 1024 + (ci - 1024)) * 3 + k];
    } else if (n < 1536) {
      int o = n - 1024;
      v = (ci < 1024) ? 0.25f * w_m1[(o * 1024 + ci) * 3 + k]
                      : 0.75f * w_m1[(o * 1024 + (ci - 1024)) * 3 + k];
    } else if (n < 2048) {
      int o = n - 1536;
      v = (ci < 1024) ? 0.75f * w_m2[(o * 1024 + ci) * 3 + k]
                      : 0.25f * w_m2[(o * 1024 + (ci - 1024)) * 3 + k];
    } else {
      int o = n - 2048;
      v = w_g1[(o * 2048 + ci) * 3 + k];
    }
    WBt[idx] = (__bf16)v;
  }
}

__global__ void aicl_pack_bias(const float* __restrict__ brgb, const float* __restrict__ bflow,
                               const float* __restrict__ bm1,  const float* __restrict__ bm2,
                               const float* __restrict__ bg1,  float* __restrict__ bias) {
  int n = blockIdx.x * blockDim.x + threadIdx.x;
  if (n >= NOUT_) return;
  float v;
  if      (n < 512)  v = brgb[n];
  else if (n < 1024) v = bflow[n - 512];
  else if (n < 1536) v = bm1[n - 1024];
  else if (n < 2048) v = bm2[n - 1536];
  else               v = bg1[n - 2048];
  bias[n] = v;
}

// ------------------------------ GEMM (WMMA) --------------------------------
#define TM  128
#define TN  128
#define TK  32
#define LDA 48   // padded LDS stride in bf16 elements (96 B, 16B aligned)
#define NSTEP 192   // 3 taps * 64 k-tiles

__device__ __forceinline__ v16bf ld_frag2_(const __bf16* p0, const __bf16* p1) {
  union { v16bf v; uint4 q[2]; } u;
  u.q[0] = *reinterpret_cast<const uint4*>(p0);
  u.q[1] = *reinterpret_cast<const uint4*>(p1);
  return u.v;
}

__global__ __launch_bounds__(256)
void aicl_gemm_wmma(const float* __restrict__ x, const __bf16* __restrict__ WBt,
                    const float* __restrict__ bias, __bf16* __restrict__ S) {
  __shared__ __align__(16) __bf16 As[2][TM * LDA];   // 2 x 12 KB
  __shared__ __align__(16) __bf16 Bs[2][TN * LDA];   // 2 x 12 KB (rows = N)

  const int tid  = threadIdx.x;
  const int lane = tid & 31;
  const int wid  = tid >> 5;
  const int wm   = (wid >> 1) * 32;
  const int wn   = (wid & 1) * 64;
  const int t0   = blockIdx.x * TM;
  const int n0   = blockIdx.y * TN;
  const int b    = blockIdx.z;

  v8f acc[2][4];
#pragma unroll
  for (int mi = 0; mi < 2; ++mi)
#pragma unroll
    for (int ni = 0; ni < 4; ++ni)
      acc[mi][ni] = (v8f){0.f, 0.f, 0.f, 0.f, 0.f, 0.f, 0.f, 0.f};

  const int arow = tid >> 1;            // 0..127
  const int acol = (tid & 1) * 16;      // 0 / 16 elems
  const int krow  = lane & 15;
  const int khalf = lane >> 4;

  // ---- A stage: f32 -> packed bf16 in VGPRs -> 2x ds_store_b128
  auto stageA = [&](int buf, int s) {
    const int tap = s >> 6;
    const int ck  = (s & 63) * TK;
    const int tin = t0 + arow + tap - 1;
    uint4 q0, q1;
    if (tin >= 0 && tin < T_) {
      const float4* p =
          reinterpret_cast<const float4*>(x + ((size_t)b * T_ + tin) * CIN_ + ck + acol);
      float4 f0 = p[0], f1 = p[1], f2 = p[2], f3 = p[3];
      if (ck + TK < CIN_) __builtin_prefetch(x + ((size_t)b * T_ + tin) * CIN_ + ck + TK + acol, 0, 1);
      q0 = make_uint4(pkbf_(f0.x, f0.y), pkbf_(f0.z, f0.w), pkbf_(f1.x, f1.y), pkbf_(f1.z, f1.w));
      q1 = make_uint4(pkbf_(f2.x, f2.y), pkbf_(f2.z, f2.w), pkbf_(f3.x, f3.y), pkbf_(f3.z, f3.w));
    } else {
      q0 = make_uint4(0u, 0u, 0u, 0u);
      q1 = q0;
    }
    uint4* d = reinterpret_cast<uint4*>(&As[buf][arow * LDA + acol]);
    d[0] = q0;
    d[1] = q1;
  };

  // ---- B stage: contiguous 8KB panel -> LDS (TDM with pad, else sync copy)
  auto stageB = [&](int buf, int s) {
    const int tap = s >> 6;
    const int ct  = s & 63;
    const __bf16* gbase = WBt + ((size_t)(tap * 64 + ct) * NOUT_ + n0) * 32;
#if __has_builtin(__builtin_amdgcn_tensor_load_to_lds) && __has_builtin(__builtin_amdgcn_s_wait_tensorcnt)
    if (wid == 0) {
      // D# : 2D tile, data_size=4B, 16 dwords/row, 128 rows, LDS pad:
      // after every 64B (pad_interval=3) insert 32B (pad_amount=7) => 96B rows.
      unsigned laddr = (unsigned)(size_t)(&Bs[buf][0]);
      unsigned long long ga = (unsigned long long)(size_t)gbase;
      u32x4_ g0;
      g0[0] = 1u;                                   // count=1 (valid descriptor)
      g0[1] = laddr;                                // lds_addr
      g0[2] = (unsigned)(ga & 0xFFFFFFFFu);         // global_addr[31:0]
      g0[3] = (unsigned)((ga >> 32) & 0x01FFFFFFu) | (2u << 30);  // addr[56:32]|type=2
      i32x8_ g1;
      g1[0] = (int)((2u << 16) | (1u << 20) | (3u << 22) | (7u << 25));
      g1[1] = (int)(16u << 16);    // tensor_dim0 = 16 dwords
      g1[2] = (int)(128u << 16);   // tensor_dim1 = 128 rows
      g1[3] = (int)(16u << 16);    // tile_dim0 = 16 dwords
      g1[4] = (int)128;            // tile_dim1 = 128 rows, tile_dim2 = 0
      g1[5] = (int)16;             // tensor_dim0_stride = 16 dwords
      g1[6] = 0;
      g1[7] = 0;
      i32x4_ z4 = {0, 0, 0, 0};
      i32x8_ z8 = {0, 0, 0, 0, 0, 0, 0, 0};
      __builtin_amdgcn_tensor_load_to_lds(g0, g1, z4, z4, z8, 0);
    }
#else
    {
      const int rowN = tid >> 1;
      const int half = tid & 1;
      const uint4* gp = reinterpret_cast<const uint4*>(gbase + rowN * 32 + half * 16);
      uint4* dp = reinterpret_cast<uint4*>(&Bs[buf][rowN * LDA + half * 16]);
      dp[0] = gp[0];
      dp[1] = gp[1];
    }
#endif
  };

  auto waitStage = [&]() {
#if __has_builtin(__builtin_amdgcn_tensor_load_to_lds) && __has_builtin(__builtin_amdgcn_s_wait_tensorcnt)
    if (wid == 0) __builtin_amdgcn_s_wait_tensorcnt(0);
#endif
  };

  stageA(0, 0);
  stageB(0, 0);
  waitStage();
  __syncthreads();

  for (int s = 0; s < NSTEP; ++s) {
    const int cur = s & 1;

    // fragment loads (documented wave32 VGPR layouts)
    v16bf afr[2], bfr[4];
#pragma unroll
    for (int mi = 0; mi < 2; ++mi) {
      const __bf16* pa = &As[cur][(wm + mi * 16 + krow) * LDA + khalf * 8];
      afr[mi] = ld_frag2_(pa, pa + 16);
    }
#pragma unroll
    for (int ni = 0; ni < 4; ++ni) {
      const __bf16* pb = &Bs[cur][(wn + ni * 16 + krow) * LDA + khalf * 8];
      bfr[ni] = ld_frag2_(pb, pb + 16);
    }

    // stage next tile into the other buffer while WMMAs run
    if (s + 1 < NSTEP) {
      stageA(1 - cur, s + 1);
      stageB(1 - cur, s + 1);
    }

#pragma unroll
    for (int mi = 0; mi < 2; ++mi)
#pragma unroll
      for (int ni = 0; ni < 4; ++ni)
        acc[mi][ni] = __builtin_amdgcn_wmma_f32_16x16x32_bf16(
            false, afr[mi], false, bfr[ni], (short)0, acc[mi][ni], false, false);

    waitStage();
    __syncthreads();
  }

  // ---- epilogue: bias + relu, bf16 store
  const int orow = (lane >> 4) * 8;
  const int ocol = lane & 15;
#pragma unroll
  for (int mi = 0; mi < 2; ++mi) {
#pragma unroll
    for (int ni = 0; ni < 4; ++ni) {
      const int cc = n0 + wn + ni * 16 + ocol;
      const float bb = bias[cc];
#pragma unroll
      for (int r = 0; r < 8; ++r) {
        const int rr = t0 + wm + mi * 16 + orow + r;
        float v = acc[mi][ni][r] + bb;
        v = v > 0.f ? v : 0.f;
        S[((size_t)b * T_ + rr) * NOUT_ + cc] = (__bf16)v;
      }
    }
  }
}

// ------------------------------ fused epilogue -----------------------------
__global__ __launch_bounds__(256)
void aicl_epilogue(const __bf16* __restrict__ S,
                   const float* wg2, const float* bg2,
                   const float* wcls, const float* bcls,
                   const float* wcrgb, const float* bcrgb,
                   const float* wcflow, const float* bcflow,
                   const float* wcm1, const float* bcm1,
                   const float* wcm2, const float* bcm2,
                   float* __restrict__ EmbC, float* __restrict__ AN1,
                   float* __restrict__ AN2, float* __restrict__ out) {
  __shared__ float row[NOUT_];
  __shared__ float red[256];
  __shared__ float embsh[512];
  __shared__ float gsm[4];
  __shared__ float heads[4];
  const int tid = threadIdx.x;
  const size_t bt = blockIdx.x;
  const int b = (int)(bt / T_);
  const int t = (int)(bt % T_);

  const __bf16* srow = S + bt * NOUT_;
  for (int j = tid; j < NOUT_; j += 256) row[j] = (float)srow[j];
  __syncthreads();

  float gl[4];
#pragma unroll
  for (int c = 0; c < 4; ++c)
    gl[c] = blockReduceSum(row[2048 + tid] * wg2[c * 512 + tid] +
                           row[2048 + 256 + tid] * wg2[c * 512 + 256 + tid], red);
  float hr = blockReduceSum(row[tid] * wcrgb[tid] + row[256 + tid] * wcrgb[256 + tid], red);
  float hf = blockReduceSum(row[512 + tid] * wcflow[tid] + row[768 + tid] * wcflow[256 + tid], red);
  float h1 = blockReduceSum(row[1024 + tid] * wcm1[tid] + row[1280 + tid] * wcm1[256 + tid], red);
  float h2 = blockReduceSum(row[1536 + tid] * wcm2[tid] + row[1792 + tid] * wcm2[256 + tid], red);

  if (tid == 0) {
    float l0 = gl[0] + bg2[0], l1 = gl[1] + bg2[1], l2 = gl[2] + bg2[2], l3 = gl[3] + bg2[3];
    float m = fmaxf(fmaxf(l0, l1), fmaxf(l2, l3));
    float e0 = __expf(l0 - m), e1 = __expf(l1 - m), e2 = __expf(l2 - m), e3 = __expf(l3 - m);
    float si = 1.f / (e0 + e1 + e2 + e3);
    gsm[0] = e0 * si; gsm[1] = e1 * si; gsm[2] = e2 * si; gsm[3] = e3 * si;
    heads[0] = sigmoidf_(hr + bcrgb[0]);
    heads[1] = sigmoidf_(hf + bcflow[0]);
    heads[2] = sigmoidf_(h1 + bcm1[0]);
    heads[3] = sigmoidf_(h2 + bcm2[0]);
  }
  __syncthreads();

  const float g0 = gsm[0], g1 = gsm[1], g2 = gsm[2], g3 = gsm[3];
  for (int i = tid; i < 512; i += 256) {
    float e = 0.5f * g0 * row[i] + (0.5f * g1 + 0.75f * g2) * row[1024 + i] +
              0.25f * g3 * row[1536 + i];
    embsh[i] = e;
    EmbC[bt * 512 + i] = e;
  }
  __syncthreads();

  float cas_sum = 0.f;
  for (int c = 0; c < NCLS; ++c) {
    float d = blockReduceSum(embsh[tid] * wcls[c * 512 + tid] +
                             embsh[256 + tid] * wcls[c * 512 + 256 + tid], red);
    if (tid == 0) {
      d += bcls[c];
      out[OFF_CAS + bt * NCLS + c] = d;
      cas_sum += d;
    }
  }
  if (tid == 0) {
    float an1 = sigmoidf_(cas_sum);
    float ar = heads[0], af = heads[1], a1 = heads[2], a2 = heads[3];
    float an2 = (0.6f * a1 + 0.4f * a2 + 0.5f * ar + 0.5f * af) * 0.25f;
    an2 = 0.8f * an2 + 0.2f * fminf(ar, af);
    AN1[bt] = an1; AN2[bt] = an2;
    out[OFF_AN1 + bt] = an1; out[OFF_AN2 + bt] = an2;
    out[OFF_AFLOW + bt] = af; out[OFF_ARGB + bt] = ar;
#pragma unroll
    for (int c = 0; c < 4; ++c) out[OFF_GATE + ((size_t)b * 4 + c) * T_ + t] = gsm[c];
  }
}

// --------------------------- percentile / max ------------------------------
__global__ __launch_bounds__(256)
void aicl_percentile(const float* __restrict__ AN1, const float* __restrict__ AN2,
                     float* __restrict__ THR, float* __restrict__ MAXA) {
  __shared__ float sv[T_];
  const int which = blockIdx.x & 1;
  const int b = blockIdx.x >> 1;
  const float* src = (which ? AN2 : AN1) + (size_t)b * T_;
  for (int i = threadIdx.x; i < T_; i += 256) sv[i] = src[i];
  __syncthreads();
  for (int k = 2; k <= T_; k <<= 1) {
    for (int j = k >> 1; j > 0; j >>= 1) {
      for (int i = threadIdx.x; i < T_; i += 256) {
        int ixj = i ^ j;
        if (ixj > i) {
          bool asc = ((i & k) == 0);
          float a = sv[i], c = sv[ixj];
          if ((a > c) == asc) { sv[i] = c; sv[ixj] = a; }
        }
      }
      __syncthreads();
    }
  }
  if (threadIdx.x == 0) {
    float q = 0.6f * (float)(T_ - 1);
    int i0 = (int)q;
    float fr = q - (float)i0;
    THR[which * B_ + b] = sv[i0] + fr * (sv[i0 + 1] - sv[i0]);
    if (which == 0) MAXA[b] = sv[T_ - 1];
  }
}

__global__ void aicl_binarize(const float* __restrict__ AN1, const float* __restrict__ AN2,
                              const float* __restrict__ THR, float* __restrict__ BIN1,
                              float* __restrict__ BIN2, float* __restrict__ out) {
  int i = blockIdx.x * blockDim.x + threadIdx.x;
  if (i >= B_ * T_) return;
  int b = i / T_;
  float b1 = AN1[i] > THR[b] ? 1.f : 0.f;
  float b2 = AN2[i] > THR[B_ + b] ? 1.f : 0.f;
  BIN1[i] = b1; BIN2[i] = b2;
  out[OFF_BIN1 + i] = b1; out[OFF_BIN2 + i] = b2;
}

// ------------------------------ top-k mining -------------------------------
__global__ __launch_bounds__(256)
void aicl_topk(const float* __restrict__ AN1, const float* __restrict__ BIN1,
               const float* __restrict__ BIN2, const float* __restrict__ MAXA,
               int* __restrict__ IDX) {
  __shared__ float sv[T_];
  __shared__ int si[T_];
  __shared__ unsigned char flags[T_];
  const int bx  = blockIdx.x;         // 192 blocks: [src(3)][cat(4)][b(16)]
  const int src = bx >> 6;
  const int cat = (bx >> 4) & 3;
  const int b   = bx & 15;
  const int ktop = (cat < 2) ? KC_TOP : KI_TOP;
  const int krnd = (cat < 2) ? KC_RND : KI_RND;
  const float maxa = MAXA[b];

  for (int t = threadIdx.x; t < T_; t += 256) {
    float an  = AN1[b * T_ + t];
    float s12 = BIN1[b * T_ + t] + BIN2[b * T_ + t];
    float sc;
    if (cat == 0)      sc = (s12 > 1.5f) ? an : 0.f;
    else if (cat == 1) sc = (s12 < 0.5f) ? (maxa - an) : 0.f;
    else {
      float inner = (s12 > 0.5f && s12 < 1.5f) ? 1.f : 0.f;
      if (cat == 2) sc = an * inner * sigmoidf_((an - 0.4f) * 5.f);
      else          sc = (maxa - an) * inner + sigmoidf_((0.3f - an) * 5.f) * inner;
    }
    sc = fminf(fmaxf(sc, -1e5f), 1e5f);
    sv[t] = sc; si[t] = t; flags[t] = 0;
  }
  __syncthreads();
  bitonicDesc_(sv, si);

  const int base = ((src * 4 + cat) * B_ + b) * KC_;
  for (int j = threadIdx.x; j < ktop; j += 256) {
    IDX[base + j] = si[j];
    flags[si[j]] = 1;
  }
  __syncthreads();

  for (int t = threadIdx.x; t < T_; t += 256) {
    sv[t] = flags[t] ? -1.f : rnd01_(0x9E37u + src, 0x85EBu + cat, b, t);
    si[t] = t;
  }
  __syncthreads();
  bitonicDesc_(sv, si);
  for (int j = threadIdx.x; j < krnd; j += 256) IDX[base + ktop + j] = si[j];
}

// ------------------------------ gather rows --------------------------------
__global__ __launch_bounds__(128)
void aicl_gather(const float* __restrict__ EmbC, const __bf16* __restrict__ S,
                 const int* __restrict__ IDX, float* __restrict__ out) {
  int r = blockIdx.x;                       // 3 * 16 * 340 rows
  const int rows_per_src = B_ * (2 * KC_ + 2 * KI_);
  const int src = r / rows_per_src; r -= src * rows_per_src;
  const int b = r / (2 * KC_ + 2 * KI_);   r -= b * (2 * KC_ + 2 * KI_);
  int cat, j, k; size_t catoff;
  if (r < KC_)               { cat = 0; j = r;                 k = KC_; catoff = 0; }
  else if (r < 2 * KC_)      { cat = 1; j = r - KC_;           k = KC_; catoff = CAT1_OFF; }
  else if (r < 2 * KC_ + KI_){ cat = 2; j = r - 2 * KC_;       k = KI_; catoff = CAT2_OFF; }
  else                       { cat = 3; j = r - 2 * KC_ - KI_; k = KI_; catoff = CAT3_OFF; }
  const int t = IDX[((src * 4 + cat) * B_ + b) * KC_ + j];
  float* dst = out + OFF_MIN + (size_t)src * SRC_STRIDE + catoff + ((size_t)b * k + j) * 512;
  if (src == 0) {
    const float* s = EmbC + ((size_t)b * T_ + t) * 512;
    for (int i = threadIdx.x; i < 512; i += 128) dst[i] = s[i];
  } else {
    const __bf16* s = S + ((size_t)b * T_ + t) * NOUT_ + (src == 1 ? 0 : 512);
    for (int i = threadIdx.x; i < 512; i += 128) dst[i] = (float)s[i];
  }
}

// ------------------------------ launcher -----------------------------------
extern "C" void kernel_launch(void* const* d_in, const int* in_sizes, int n_in,
                              void* d_out, int out_size, void* d_ws, size_t ws_size,
                              hipStream_t stream) {
  (void)in_sizes; (void)n_in; (void)out_size; (void)ws_size;
  const float* x      = (const float*)d_in[0];
  const float* w_rgb  = (const float*)d_in[1];
  const float* b_rgb  = (const float*)d_in[2];
  const float* w_crgb = (const float*)d_in[3];
  const float* b_crgb = (const float*)d_in[4];
  const float* w_flow = (const float*)d_in[5];
  const float* b_flow = (const float*)d_in[6];
  const float* w_cflow= (const float*)d_in[7];
  const float* b_cflow= (const float*)d_in[8];
  const float* w_m1   = (const float*)d_in[9];
  const float* b_m1   = (const float*)d_in[10];
  const float* w_cm1  = (const float*)d_in[11];
  const float* b_cm1  = (const float*)d_in[12];
  const float* w_m2   = (const float*)d_in[13];
  const float* b_m2   = (const float*)d_in[14];
  const float* w_cm2  = (const float*)d_in[15];
  const float* b_cm2  = (const float*)d_in[16];
  const float* w_g1   = (const float*)d_in[17];
  const float* b_g1   = (const float*)d_in[18];
  const float* w_g2   = (const float*)d_in[19];
  const float* b_g2   = (const float*)d_in[20];
  const float* w_cls  = (const float*)d_in[21];
  const float* b_cls  = (const float*)d_in[22];
  float* out = (float*)d_out;

  char* ws = (char*)d_ws;
  size_t off = 0;
  auto alloc = [&](size_t bytes) { size_t o = off; off = (off + bytes + 255) & ~(size_t)255; return o; };
  __bf16* WBt  = (__bf16*)(ws + alloc((size_t)3 * CIN_ * NOUT_ * 2));   // 31.5 MB
  float*  bias = (float*)(ws + alloc((size_t)NOUT_ * 4));
  __bf16* S    = (__bf16*)(ws + alloc((size_t)B_ * T_ * NOUT_ * 2));    // 168 MB
  float*  EmbC = (float*)(ws + alloc((size_t)B_ * T_ * 512 * 4));       // 67 MB
  float*  AN1  = (float*)(ws + alloc((size_t)B_ * T_ * 4));
  float*  AN2  = (float*)(ws + alloc((size_t)B_ * T_ * 4));
  float*  THR  = (float*)(ws + alloc((size_t)2 * B_ * 4));
  float*  MAXA = (float*)(ws + alloc((size_t)B_ * 4));
  float*  BIN1 = (float*)(ws + alloc((size_t)B_ * T_ * 4));
  float*  BIN2 = (float*)(ws + alloc((size_t)B_ * T_ * 4));
  int*    IDX  = (int*)(ws + alloc((size_t)3 * 4 * B_ * KC_ * 4));

  aicl_pack_w<<<8192, 256, 0, stream>>>(w_rgb, w_flow, w_m1, w_m2, w_g1, WBt);
  aicl_pack_bias<<<(NOUT_ + 255) / 256, 256, 0, stream>>>(b_rgb, b_flow, b_m1, b_m2, b_g1, bias);

  dim3 gemm_grid(T_ / TM, NOUT_ / TN, B_);
  aicl_gemm_wmma<<<gemm_grid, 256, 0, stream>>>(x, WBt, bias, S);

  aicl_epilogue<<<B_ * T_, 256, 0, stream>>>(S, w_g2, b_g2, w_cls, b_cls,
                                             w_crgb, b_crgb, w_cflow, b_cflow,
                                             w_cm1, b_cm1, w_cm2, b_cm2,
                                             EmbC, AN1, AN2, out);

  aicl_percentile<<<2 * B_, 256, 0, stream>>>(AN1, AN2, THR, MAXA);
  aicl_binarize<<<(B_ * T_ + 255) / 256, 256, 0, stream>>>(AN1, AN2, THR, BIN1, BIN2, out);
  aicl_topk<<<3 * 4 * B_, 256, 0, stream>>>(AN1, BIN1, BIN2, MAXA, IDX);
  aicl_gather<<<3 * B_ * (2 * KC_ + 2 * KI_), 128, 0, stream>>>(EmbC, S, IDX, out);
}
